// SimCLRLoss_16131897164009
// MI455X (gfx1250) — matrix operand order repaired and checked by reading, not compiled, not run on hardware
//
#include <hip/hip_runtime.h>
#include <hip/hip_bf16.h>

// ---------------------------------------------------------------------------
// SimCLR loss, fused flash-logsumexp over f16 WMMA tiles (gfx1250 / CDNA5).
//   features: 8192 x 128 f32  ->  scalar loss (f32)
// Pipeline: normalize->f16 (ws) | fused fn*fn^T via v_wmma_f32_16x16x32_f16
//   (register double-buffered B tiles, split-K dual accumulators, scalar
//   uniform control flow) + bounded-max logsumexp | deterministic mean.
// ---------------------------------------------------------------------------

typedef _Float16 h4  __attribute__((ext_vector_type(4)));
typedef _Float16 h8  __attribute__((ext_vector_type(8)));
typedef _Float16 h16 __attribute__((ext_vector_type(16)));
typedef float    v8f __attribute__((ext_vector_type(8)));

#define N_ROWS   8192
#define K_DIM    128
#define HALF_N   4096
#define NT       (N_ROWS / 16)                 /* 512 column tiles           */
#define INV_T    14.285714285714286f           /* 1 / 0.07                   */
#define LOG2E    1.4426950408889634f
#define LN2      0.6931471805599453f
#define B2       (INV_T * LOG2E)               /* logit upper bound, base-2  */
#define S2SCALE  (INV_T * LOG2E)

// ---------------------------------------------------------------------------
// Kernel 1: L2-normalize rows, emit f16 matrix into workspace.
// One wave per row (32 lanes x float4 = 128 elements).
// ---------------------------------------------------------------------------
__global__ __launch_bounds__(256) void simclr_normalize_f16(
    const float* __restrict__ f, _Float16* __restrict__ fh) {
  const int lane = threadIdx.x & 31;
  const int row  = blockIdx.x * 8 + (threadIdx.x >> 5);

  const float4 x = ((const float4*)(f + row * K_DIM))[lane];
  float ss = x.x * x.x + x.y * x.y + x.z * x.z + x.w * x.w;
#pragma unroll
  for (int o = 16; o > 0; o >>= 1) ss += __shfl_xor(ss, o, 32);
  const float inv = rsqrtf(ss);

  h4 y;
  y[0] = (_Float16)(x.x * inv);
  y[1] = (_Float16)(x.y * inv);
  y[2] = (_Float16)(x.z * inv);
  y[3] = (_Float16)(x.w * inv);
  *(h4*)(fh + row * K_DIM + lane * 4) = y;
}

// A-tile fragment (16x32 f16, MxK): lane L -> row L%16,
// K = {kc+kh .. +7} in regs 0..7 and {kc+16+kh ..} in regs 8..15,
// kh = 0 for lanes 0-15, 8 for lanes 16-31.  (ISA 7.12.2)
__device__ inline h16 load_a_k32(const _Float16* __restrict__ rowp,
                                 int kc, int kh) {
  h8 lo = *(const h8*)(rowp + kc + kh);
  h8 hi = *(const h8*)(rowp + kc + 16 + kh);
  h16 r;
#pragma unroll
  for (int i = 0; i < 8; ++i) { r[i] = lo[i]; r[i + 8] = hi[i]; }
  return r;
}

// B-tile fragment (32x16 f16, KxN): lane L -> column L%16 (= row of fn since
// B = fn^T), same K order as A: kb = 0 (lanes 0-15) / 8 (lanes 16-31).
__device__ inline h16 load_b_k32(const _Float16* __restrict__ rowp,
                                 int kc, int kb) {
  return load_a_k32(rowp, kc, kb);
}

// ---------------------------------------------------------------------------
// Kernel 2: fused similarity + bounded-max logsumexp.
// Block b owns rows 16b..16b+15; 8 waves each stream 64 of the 512 col-tiles
// with a 2-deep register pipeline on the B tiles.
// ---------------------------------------------------------------------------
__global__ __launch_bounds__(256) void simclr_fused_lse(
    const _Float16* __restrict__ fh, float* __restrict__ rowloss) {
  const int lane  = threadIdx.x & 31;
  // SGPR wave id -> all tile-loop control flow becomes scalar branches.
  const int wave  = __builtin_amdgcn_readfirstlane(threadIdx.x >> 5);
  const int l16   = lane & 15;
  const int hi    = (lane >= 16);
  const int kh    = hi ? 8 : 0;    // A fragment K-half select
  const int kb    = hi ? 8 : 0;    // B fragment K-half select
  const int rbase = blockIdx.x * 16;

  // Only these two column tiles ever contain the diagonal / positive entry
  // for this block's 16 rows -> masks live behind scalar branches.
  const int diag_tile = blockIdx.x;
  const int pos_tile  = (blockIdx.x + HALF_N / 16) & (NT - 1);
  const int myrow     = l16;       // in-tile column matching row v+8*hi

  // Resident A tile: rows rbase..rbase+15, K = 0..127 (4 x v16h).
  const _Float16* arow = fh + (rbase + l16) * K_DIM;
  h16 a[4];
#pragma unroll
  for (int c = 0; c < 4; ++c) a[c] = load_a_k32(arow, 32 * c, kh);

  float sum[8];
  float posv[8];
#pragma unroll
  for (int v = 0; v < 8; ++v) { sum[v] = 0.0f; posv[v] = -1.0e30f; }

  // Prime the B pipeline.
  h16 b[4];
  {
    const _Float16* brow = fh + (wave * 16 + l16) * K_DIM;
#pragma unroll
    for (int c = 0; c < 4; ++c) b[c] = load_b_k32(brow, 32 * c, kb);
  }

  for (int jt = wave; jt < NT; jt += 8) {
    const int jn = jt + 8;
    h16 bn[4];
    if (jn < NT) {  // scalar branch: issue next tile's loads first
      const _Float16* brow = fh + (jn * 16 + l16) * K_DIM;
#pragma unroll
      for (int c = 0; c < 4; ++c) bn[c] = load_b_k32(brow, 32 * c, kb);
    }

    // Split-K: two independent accumulator chains to break the WMMA RAW
    // dependency (each chain is 2 deep instead of 4).
    v8f acc0 = {}, acc1 = {};
    acc0 = __builtin_amdgcn_wmma_f32_16x16x32_f16(false, a[0], false, b[0],
                                                  (short)0, acc0, false, false);
    acc1 = __builtin_amdgcn_wmma_f32_16x16x32_f16(false, a[1], false, b[1],
                                                  (short)0, acc1, false, false);
    acc0 = __builtin_amdgcn_wmma_f32_16x16x32_f16(false, a[2], false, b[2],
                                                  (short)0, acc0, false, false);
    acc1 = __builtin_amdgcn_wmma_f32_16x16x32_f16(false, a[3], false, b[3],
                                                  (short)0, acc1, false, false);
    v8f acc;
#pragma unroll
    for (int v = 0; v < 8; ++v) acc[v] = acc0[v] + acc1[v];

    // C layout: reg v, lanes 0-15 -> M=v ; lanes 16-31 -> M=v+8 ; N = lane%16.
    // Element (v, lane): row = rbase+v+8*hi, col = 16*jt + l16.
    if (jt == diag_tile) {             // scalar-rare: mask diagonal element
#pragma unroll
      for (int v = 0; v < 8; ++v) {
        const float e = __builtin_amdgcn_exp2f(fmaf(acc[v], S2SCALE, -B2));
        sum[v] += (myrow == v + 8 * hi) ? 0.0f : e;
      }
    } else {
      if (jt == pos_tile) {            // scalar-rare: capture positive logit
#pragma unroll
        for (int v = 0; v < 8; ++v)
          if (myrow == v + 8 * hi) posv[v] = acc[v] * INV_T;
      }
#pragma unroll                         // hot path: fma + v_exp_f32 + add
      for (int v = 0; v < 8; ++v)
        sum[v] += __builtin_amdgcn_exp2f(fmaf(acc[v], S2SCALE, -B2));
    }

    if (jn < NT) {
#pragma unroll
      for (int c = 0; c < 4; ++c) b[c] = bn[c];
    }
  }

  __shared__ float s_sum[8 * 8 * 32];
  __shared__ float s_pos[8 * 8 * 32];
#pragma unroll
  for (int v = 0; v < 8; ++v) {
    s_sum[(wave * 8 + v) * 32 + lane] = sum[v];
    s_pos[(wave * 8 + v) * 32 + lane] = posv[v];
  }
  __syncthreads();

  if (threadIdx.x < 16) {
    const int r  = threadIdx.x;
    const int v  = r & 7;
    const int l0 = (r >> 3) * 16;  // rows 0-7 in lanes 0-15, rows 8-15 in 16-31
    float tot = 0.0f, p = -1.0e30f;
    for (int w = 0; w < 8; ++w)
#pragma unroll
      for (int l = 0; l < 16; ++l) {
        const int idx = (w * 8 + v) * 32 + l0 + l;
        tot += s_sum[idx];
        p = fmaxf(p, s_pos[idx]);
      }
    const float lse = (B2 + log2f(tot)) * LN2;
    rowloss[rbase + r] = lse - p;
  }
}

// ---------------------------------------------------------------------------
// Kernel 3: deterministic mean over 8192 per-row losses (single block).
// ---------------------------------------------------------------------------
__global__ __launch_bounds__(256) void simclr_mean(
    const float* __restrict__ rowloss, float* __restrict__ out) {
  __shared__ float buf[256];
  float s = 0.0f;
  for (int i = threadIdx.x; i < N_ROWS; i += 256) s += rowloss[i];
  buf[threadIdx.x] = s;
  __syncthreads();
  for (int o = 128; o > 0; o >>= 1) {
    if (threadIdx.x < o) buf[threadIdx.x] += buf[threadIdx.x + o];
    __syncthreads();
  }
  if (threadIdx.x == 0) out[0] = buf[0] * (1.0f / (float)N_ROWS);
}

// ---------------------------------------------------------------------------
extern "C" void kernel_launch(void* const* d_in, const int* in_sizes, int n_in,
                              void* d_out, int out_size, void* d_ws,
                              size_t ws_size, hipStream_t stream) {
  (void)in_sizes; (void)n_in; (void)out_size; (void)ws_size;
  const float* feat = (const float*)d_in[0];
  float* out = (float*)d_out;

  _Float16* fh      = (_Float16*)d_ws;                         // 2 MB
  float*    rowloss = (float*)((char*)d_ws +
                               (size_t)N_ROWS * K_DIM * sizeof(_Float16));

  simclr_normalize_f16<<<N_ROWS / 8, 256, 0, stream>>>(feat, fh);
  simclr_fused_lse<<<N_ROWS / 16, 256, 0, stream>>>(fh, rowloss);
  simclr_mean<<<1, 256, 0, stream>>>(rowloss, out);
}
